// LinearSGD_63170378990326
// MI455X (gfx1250) — compile-verified, matrix operand order checked by reading, not run on hardware
//
#include <hip/hip_runtime.h>
#include <hip/hip_bf16.h>
#include <math.h>
#include <stdint.h>

// ---------------------------------------------------------------------------
// Problem constants (match reference)
// ---------------------------------------------------------------------------
#define H_  16
#define K_  128
#define V_  128
#define BT_ 64
#define B_  2
#define T_  4096
#define D_  2048
#define NT_ (T_ / BT_)
#define BASE_LR_ 0.1f
#define EPS_ 1e-5f

// Swizzled-tile row stride (bf16 elements) for one 32-wide K chunk (+8 pad)
#define SWS_ 40

typedef __attribute__((ext_vector_type(16))) __bf16 v16bf;
typedef __attribute__((ext_vector_type(8)))  __bf16 bf8v;
typedef __attribute__((ext_vector_type(2)))  __bf16 bf2v;
typedef __attribute__((ext_vector_type(8)))  float  v8f;
typedef __attribute__((ext_vector_type(4)))  unsigned int u32x4;
typedef __attribute__((ext_vector_type(8)))  int i32x8;
typedef __attribute__((ext_vector_type(4)))  int i32x4;

#if __has_builtin(__builtin_amdgcn_tensor_load_to_lds) && \
    __has_builtin(__builtin_amdgcn_tensor_store_from_lds) && \
    __has_builtin(__builtin_amdgcn_s_wait_tensorcnt)
#define HAVE_TDM 1
#else
#define HAVE_TDM 0
#endif

__device__ __forceinline__ v8f wmma_bf16(v16bf a, v16bf b, v8f c) {
  // D = A(16x32 bf16) * B(32x16 bf16) + C(16x16 f32)
  return __builtin_amdgcn_wmma_f32_16x16x32_bf16(false, a, false, b, (short)0, c,
                                                 false, false);
}

// Fragment = two contiguous ds_load_b128 from a swizzled tile row.
__device__ __forceinline__ v16bf frag_load(const __bf16* p) {
  bf8v lo = *(const bf8v*)p;
  bf8v hi = *(const bf8v*)(p + 8);
  v16bf f;
#pragma unroll
  for (int j = 0; j < 8; ++j) { f[j] = lo[j]; f[j + 8] = hi[j]; }
  return f;
}

// slot(K) within a 32-K swizzled row: ISA 16-bit fragment permutation
__device__ __forceinline__ int kslot(int kg) {
  return ((kg & 8) << 1) | (kg & 7) | ((kg & 16) >> 1);
}

#if HAVE_TDM
// Tensor-DMA descriptor group 0: count=1 user D#, LDS byte addr, 57-bit global
// byte addr, type=2 ("image") in bits 127:126.
__device__ __forceinline__ u32x4 tdm_group0(unsigned lds_off,
                                            unsigned long long gaddr) {
  u32x4 g;
  g[0] = 1u;
  g[1] = lds_off;
  g[2] = (unsigned)gaddr;
  g[3] = (unsigned)((gaddr >> 32) & 0x01ffffffu) | (2u << 30);
  return g;
}
// Group 1 for a 64-row x 128-col f32 tile, row stride 128 elements:
// data_size=4B(code 2), tensor_dim0=128, tensor_dim1=64, tile_dim0=128,
// tile_dim1=64, tensor_dim0_stride=128; no multicast/pad/iterate.
__device__ __forceinline__ i32x8 tdm_group1_64x128_f32() {
  i32x8 g;
  g[0] = (int)(2u << 16);    // workgroup_mask=0 | data_size=2 (4B)
  g[1] = (int)(128u << 16);  // tensor_dim0[15:0] at bits 63:48
  g[2] = (int)(64u << 16);   // tensor_dim0 hi=0 | tensor_dim1[15:0]
  g[3] = (int)(128u << 16);  // tensor_dim1 hi=0 | tile_dim0=128
  g[4] = 64;                 // tile_dim1=64 | tile_dim2=0
  g[5] = 128;                // tensor_dim0_stride[31:0]
  g[6] = 0;                  // stride hi | tensor_dim1_stride lo
  g[7] = 0;
  return g;
}
#endif

// ---------------------------------------------------------------------------
// f32 -> bf16 conversion (grid-stride)
// ---------------------------------------------------------------------------
__global__ void cvt_f32_bf16(const float* __restrict__ in,
                             __bf16* __restrict__ out, long long n) {
  long long i = (long long)blockIdx.x * blockDim.x + threadIdx.x;
  long long stride = (long long)gridDim.x * blockDim.x;
  for (; i < n; i += stride) out[i] = (__bf16)in[i];
}

// ---------------------------------------------------------------------------
// Tiled bf16 WMMA GEMM:  C[M,N] = scale * A[M,Kd] * B[N,Kd]^T
// 256 threads = 8 wave32 in a 4x2 grid; 128x128 block tile; each wave owns
// 2x4 16x16 subtiles (8 WMMA per k-chunk). Double-buffered swizzled LDS
// tiles; one barrier per 32-wide k-chunk.
// OUTMODE 0: C row-major [M,N]
// OUTMODE 1: head layout [B,H,T,128]  (n -> (h, d), m -> (b, t))
// ---------------------------------------------------------------------------
template <int OUTMODE>
__global__ __launch_bounds__(256) void gemm_bf16_128x128(
    const __bf16* __restrict__ A, const __bf16* __restrict__ Bw,
    float* __restrict__ C, int M, int N, int Kd, float scale) {
  __shared__ __bf16 sA[2][128][SWS_];
  __shared__ __bf16 sB[2][128][SWS_];

  const int tid  = threadIdx.x;
  const int wave = tid >> 5;
  const int lane = tid & 31;
  const int half = lane >> 4;
  const int mr   = lane & 15;
  const int bm = blockIdx.y * 128;
  const int bn = blockIdx.x * 128;

  const int wm = (wave >> 1) * 32;
  const int wn = (wave & 1) * 64;

  const int rowA  = tid >> 2;       // 0..63
  const int colg  = (tid & 3) * 8;  // 0,8,16,24
  const int slotb = ((colg & 8) << 1) | ((colg & 16) >> 1);

  v8f acc[2][4];
#pragma unroll
  for (int i = 0; i < 2; ++i)
#pragma unroll
    for (int j = 0; j < 4; ++j)
#pragma unroll
      for (int r = 0; r < 8; ++r) acc[i][j][r] = 0.f;

  const int nK = Kd >> 5;
  const __bf16* apt = A  + (size_t)(bm + rowA) * Kd + colg;
  const __bf16* bpt = Bw + (size_t)(bn + rowA) * Kd + colg;
  const size_t rstep = (size_t)64 * Kd;

  bf8v ra0 = *(const bf8v*)apt;
  bf8v ra1 = *(const bf8v*)(apt + rstep);
  bf8v rb0 = *(const bf8v*)bpt;
  bf8v rb1 = *(const bf8v*)(bpt + rstep);
  *(bf8v*)&sA[0][rowA][slotb]      = ra0;
  *(bf8v*)&sA[0][rowA + 64][slotb] = ra1;
  *(bf8v*)&sB[0][rowA][slotb]      = rb0;
  *(bf8v*)&sB[0][rowA + 64][slotb] = rb1;

  int buf = 0;
  for (int kt = 0; kt < nK; ++kt) {
    __syncthreads();
    const bool more = (kt + 1 < nK);
    if (more) {
      const __bf16* ap = apt + (kt + 1) * 32;
      const __bf16* bp = bpt + (kt + 1) * 32;
      if (kt + 2 < nK) {  // gfx1250 global_prefetch_b8
        __builtin_prefetch(ap + 32, 0, 1);
        __builtin_prefetch(bp + 32, 0, 1);
      }
      ra0 = *(const bf8v*)ap;
      ra1 = *(const bf8v*)(ap + rstep);
      rb0 = *(const bf8v*)bp;
      rb1 = *(const bf8v*)(bp + rstep);
    }
    v16bf afr[2], bfr[4];
#pragma unroll
    for (int i = 0; i < 2; ++i)
      afr[i] = frag_load(&sA[buf][wm + i * 16 + mr][half * 16]);
#pragma unroll
    for (int j = 0; j < 4; ++j)
      bfr[j] = frag_load(&sB[buf][wn + j * 16 + mr][half * 16]);
#pragma unroll
    for (int i = 0; i < 2; ++i)
#pragma unroll
      for (int j = 0; j < 4; ++j)
        acc[i][j] = wmma_bf16(afr[i], bfr[j], acc[i][j]);

    if (more) {
      buf ^= 1;
      *(bf8v*)&sA[buf][rowA][slotb]      = ra0;
      *(bf8v*)&sA[buf][rowA + 64][slotb] = ra1;
      *(bf8v*)&sB[buf][rowA][slotb]      = rb0;
      *(bf8v*)&sB[buf][rowA + 64][slotb] = rb1;
    }
  }

#pragma unroll
  for (int i = 0; i < 2; ++i)
#pragma unroll
    for (int j = 0; j < 4; ++j)
#pragma unroll
      for (int r = 0; r < 8; ++r) {
        int m = bm + wm + i * 16 + 8 * half + r;
        int n = bn + wn + j * 16 + mr;
        float val = acc[i][j][r] * scale;
        if (OUTMODE == 0) {
          C[(size_t)m * N + n] = val;
        } else {
          int b = m >> 12, t = m & (T_ - 1);
          int h = n >> 7, d = n & 127;
          C[((((size_t)b * H_ + h) * T_ + t) << 7) + d] = val;
        }
      }
}

// ---------------------------------------------------------------------------
// eta / theta projections (tiny: N = 16 per head-matrix) + sigmoid; [B,H,T]
// ---------------------------------------------------------------------------
__global__ void etatheta_kernel(const float* __restrict__ X,
                                const float* __restrict__ Weta,
                                const float* __restrict__ Wth,
                                float* __restrict__ eta,
                                float* __restrict__ theta) {
  int gid = blockIdx.x * blockDim.x + threadIdx.x;  // over B*T*H
  if (gid >= B_ * T_ * H_) return;
  int h = gid & (H_ - 1);
  int m = gid >> 4;  // b*T + t
  int b = m >> 12, t = m & (T_ - 1);
  const float* x  = X + (size_t)m * D_;
  const float* we = Weta + (size_t)h * D_;
  const float* wt = Wth + (size_t)h * D_;
  float se = 0.f, st = 0.f;
  for (int d = 0; d < D_; ++d) {
    float xv = x[d];
    se += xv * we[d];
    st += xv * wt[d];
  }
  size_t oi = ((size_t)b * H_ + h) * T_ + t;
  eta[oi]   = 1.f / (1.f + expf(-se));
  theta[oi] = BASE_LR_ * (1.f / (1.f + expf(-st)));
}

// ---------------------------------------------------------------------------
// Cooperative LDS GEMM used by the scan kernel (templated -> fully unrolled).
// C[M,N] (+)= A[M,KD] * B[N,KD]^T, f32 operands in LDS.
// Staging converts+swizzles to bf16 (paired b32 stores); 8 waves in a 4x2
// grid, each owning (M/64)x(N/32) subtiles.  OP 0: C = A*B  OP 2: C -= A*B
// ---------------------------------------------------------------------------
template <int M, int N, int KD, int OP>
__device__ void lds_gemm(const float* A, int rsA, int csA,
                         const float* mScaleA, const float* kScaleA,
                         const float* B, int rsB, int csB,
                         float* C, int ldc, __bf16* swA, __bf16* swB) {
  const int tid  = threadIdx.x;
  const int wave = tid >> 5;
  const int lane = tid & 31;
  const int half = lane >> 4;
  const int mr   = lane & 15;
  constexpr int WM = 4, WN = 2;
  constexpr int TM = M / 16 / WM;
  constexpr int TN = N / 16 / WN;
  constexpr int NK = KD / 32;
  constexpr int KD2 = KD / 2;
  constexpr int chA = M * SWS_;
  constexpr int chB = N * SWS_;

  // ---- stage A (scaled), bf16 pairs ----
  for (int idx = tid; idx < M * KD2; idx += 256) {
    int m  = idx / KD2;
    int kg = (idx % KD2) * 2;
    float a0 = A[m * rsA + kg * csA];
    float a1 = A[m * rsA + (kg + 1) * csA];
    if (mScaleA) { float s = mScaleA[m]; a0 *= s; a1 *= s; }
    if (kScaleA) { a0 *= kScaleA[kg]; a1 *= kScaleA[kg + 1]; }
    bf2v p; p[0] = (__bf16)a0; p[1] = (__bf16)a1;
    *(bf2v*)(swA + (kg >> 5) * chA + m * SWS_ + kslot(kg & 31)) = p;
  }
  // ---- stage B ----
  for (int idx = tid; idx < N * KD2; idx += 256) {
    int n  = idx / KD2;
    int kg = (idx % KD2) * 2;
    bf2v p;
    p[0] = (__bf16)B[n * rsB + kg * csB];
    p[1] = (__bf16)B[n * rsB + (kg + 1) * csB];
    *(bf2v*)(swB + (kg >> 5) * chB + n * SWS_ + kslot(kg & 31)) = p;
  }
  __syncthreads();

  const int wm = (wave >> 1) * (TM * 16);
  const int wn = (wave & 1) * (TN * 16);

  v8f acc[TM][TN];
#pragma unroll
  for (int i = 0; i < TM; ++i)
#pragma unroll
    for (int j = 0; j < TN; ++j)
#pragma unroll
      for (int r = 0; r < 8; ++r) acc[i][j][r] = 0.f;

#pragma unroll
  for (int kc = 0; kc < NK; ++kc) {
    v16bf afr[TM], bfr[TN];
#pragma unroll
    for (int i = 0; i < TM; ++i)
      afr[i] = frag_load(swA + kc * chA + (wm + i * 16 + mr) * SWS_ + half * 16);
#pragma unroll
    for (int j = 0; j < TN; ++j)
      bfr[j] = frag_load(swB + kc * chB + (wn + j * 16 + mr) * SWS_ + half * 16);
#pragma unroll
    for (int i = 0; i < TM; ++i)
#pragma unroll
      for (int j = 0; j < TN; ++j)
        acc[i][j] = wmma_bf16(afr[i], bfr[j], acc[i][j]);
  }

#pragma unroll
  for (int i = 0; i < TM; ++i)
#pragma unroll
    for (int j = 0; j < TN; ++j)
#pragma unroll
      for (int r = 0; r < 8; ++r) {
        int m = wm + i * 16 + 8 * half + r;
        int n = wn + j * 16 + mr;
        float* cp = C + m * ldc + n;
        if (OP == 0) *cp = acc[i][j][r];
        else         *cp -= acc[i][j][r];
      }
  __syncthreads();  // swA/swB and C reused by the next phase
}

// ---------------------------------------------------------------------------
// Chunked fast-weight scan. One block per (b,h); W state (128x128 f32), all
// chunk tiles, and two swizzled bf16 scratch buffers live in LDS (~289 KB).
// q/k chunk tiles arrive via the Tensor Data Mover (TENSORcnt); z leaves via
// async tensor store overlapped with the W-update phase.
// q/k/v in [B,H,T,128]; eta/theta in [B,H,T]; z out [B,H,T,128].
// ---------------------------------------------------------------------------
__global__ __launch_bounds__(256) void scan_kernel(
    const float* __restrict__ q, const float* __restrict__ k,
    const float* __restrict__ v, const float* __restrict__ eta,
    const float* __restrict__ theta, float* __restrict__ zout) {
  extern __shared__ char smem[];
  float* Wst  = (float*)smem;         // 128*128 fast-weight state
  float* qc   = Wst + 128 * 128;      // 64*128
  float* kc   = qc + 64 * 128;        // 64*128
  float* the  = kc + 64 * 128;        // 64*128  e -> theta_e
  float* zb   = the + 64 * 128;       // 64*128  z accumulator
  float* qkb  = zb + 64 * 128;        // 64*64   qk -> qk*beta
  float* lec  = qkb + 64 * 64;        // 64  cumulative log decay
  float* tcv  = lec + 64;             // 64  theta
  float* elec = tcv + 64;             // 64  exp(lec)
  float* blast= elec + 64;            // 64  beta last row
  __bf16* swA = (__bf16*)(blast + 64);        // 4*128*SWS_ bf16 swizzle scratch
  __bf16* swB = swA + 4 * 128 * SWS_;

  const int tid = threadIdx.x;
  const int bh  = blockIdx.x;  // b*H + h

  for (int i = tid; i < 128 * 128 / 4; i += 256) ((float4*)Wst)[i] = float4{0,0,0,0};
  __syncthreads();

  for (int c = 0; c < NT_; ++c) {
    const size_t base = ((size_t)bh * T_ + (size_t)c * BT_) * 128;
    const float4* v4 = (const float4*)(v + base);

#if HAVE_TDM
    if (tid < 32) {  // wave 0 drives the DMA; wait also drains prior z store
      i32x8 g1 = tdm_group1_64x128_f32();
      i32x4 gz = {0, 0, 0, 0};
      i32x8 gpad = {0, 0, 0, 0, 0, 0, 0, 0};
      u32x4 gq = tdm_group0((unsigned)(uintptr_t)qc,
                            (unsigned long long)(uintptr_t)(q + base));
      u32x4 gk = tdm_group0((unsigned)(uintptr_t)kc,
                            (unsigned long long)(uintptr_t)(k + base));
      __builtin_amdgcn_tensor_load_to_lds(gq, g1, gz, gz, gpad, 0);
      __builtin_amdgcn_tensor_load_to_lds(gk, g1, gz, gz, gpad, 0);
      __builtin_amdgcn_s_wait_tensorcnt(0);
    }
#else
    const float4* q4 = (const float4*)(q + base);
    const float4* k4 = (const float4*)(k + base);
    for (int i = tid; i < 64 * 128 / 4; i += 256) {
      ((float4*)qc)[i] = q4[i];
      ((float4*)kc)[i] = k4[i];
    }
#endif
    if (tid < 64) {
      float ev = eta[(size_t)bh * T_ + c * BT_ + tid];
      lec[tid] = logf(fmaxf(ev, 1e-8f));
      tcv[tid] = theta[(size_t)bh * T_ + c * BT_ + tid];
    }
    __syncthreads();
    if (tid == 0) {  // serial cumsum over 64 (cheap)
      float s = 0.f;
      for (int t = 0; t < 64; ++t) { s += lec[t]; lec[t] = s; }
    }
    __syncthreads();
    if (tid < 64) {
      elec[tid]  = expf(lec[tid]);
      blast[tid] = expf(fminf(lec[63] - lec[tid], 0.f));
    }
    __syncthreads();

    // e = kc @ W^T                        (64x128x128)
    lds_gemm<64, 128, 128, 0>(kc, 128, 1, nullptr, nullptr, Wst, 128, 1,
                              the, 128, swA, swB);
    // theta_e = theta * (e - v)
    for (int i = tid; i < 64 * 128 / 4; i += 256) {
      int t = i >> 5;
      float4 vv = v4[i];
      float4 ee = ((float4*)the)[i];
      float tc = tcv[t];
      ee.x = tc * (ee.x - vv.x); ee.y = tc * (ee.y - vv.y);
      ee.z = tc * (ee.z - vv.z); ee.w = tc * (ee.w - vv.w);
      ((float4*)the)[i] = ee;
    }
    __syncthreads();
    // qk = qc @ kc^T                      (64x64x128)
    lds_gemm<64, 64, 128, 0>(qc, 128, 1, nullptr, nullptr, kc, 128, 1,
                             qkb, 64, swA, swB);
    // qk *= beta (causal decay mask)
    for (int i = tid; i < 64 * 64; i += 256) {
      int ii = i >> 6, jj = i & 63;
      qkb[i] = (jj <= ii) ? qkb[i] * expf(fminf(lec[ii] - lec[jj], 0.f)) : 0.f;
    }
    __syncthreads();
    // z = (qc * exp(lec)) @ W^T           (64x128x128, row-scaled A)
    lds_gemm<64, 128, 128, 0>(qc, 128, 1, elec, nullptr, Wst, 128, 1,
                              zb, 128, swA, swB);
    // z -= (qk*beta) @ theta_e            (64x128x64, B = theta_e^T view)
    lds_gemm<64, 128, 64, 2>(qkb, 64, 1, nullptr, nullptr, the, 1, 128,
                             zb, 128, swA, swB);
#if HAVE_TDM
    if (tid < 32) {  // async z store; overlaps W-update, drained at loop top
      i32x8 g1 = tdm_group1_64x128_f32();
      i32x4 gzero = {0, 0, 0, 0};
      i32x8 gpad = {0, 0, 0, 0, 0, 0, 0, 0};
      u32x4 gzd = tdm_group0((unsigned)(uintptr_t)zb,
                             (unsigned long long)(uintptr_t)(zout + base));
      __builtin_amdgcn_tensor_store_from_lds(gzd, g1, gzero, gzero, gpad, 0);
    }
#else
    float4* zo4 = (float4*)(zout + base);
    for (int i = tid; i < 64 * 128 / 4; i += 256) zo4[i] = ((float4*)zb)[i];
#endif
    // W = W * exp(lec[63]) - e_decay^T @ kc
    float e63 = expf(lec[63]);
    for (int i = tid; i < 128 * 128 / 4; i += 256) {
      float4 w = ((float4*)Wst)[i];
      w.x *= e63; w.y *= e63; w.z *= e63; w.w *= e63;
      ((float4*)Wst)[i] = w;
    }
    __syncthreads();
    // A = theta_e^T scaled by beta_last over k(=t); B = kc^T view
    lds_gemm<128, 128, 64, 2>(the, 1, 128, nullptr, blast, kc, 1, 128,
                              Wst, 128, swA, swB);
  }
  // S_ENDPGM implies wait-idle: final tensor store completes before exit.
}

// ---------------------------------------------------------------------------
// Gated RMS norm: y = rmsnorm(z) * norm_w * silu(g), output bf16 for final GEMM
// ---------------------------------------------------------------------------
__global__ __launch_bounds__(128) void normgate_kernel(
    const float* __restrict__ z, const float* __restrict__ g,
    const float* __restrict__ normw, __bf16* __restrict__ y) {
  __shared__ float red[128];
  const int vi  = threadIdx.x;
  const int gid = blockIdx.x;  // m*H + h
  const int h = gid & (H_ - 1);
  const int m = gid >> 4;      // b*T + t
  const int b = m >> 12, t = m & (T_ - 1);
  const size_t zi = ((((size_t)b * H_ + h) * T_ + t) << 7) + vi;
  float zv = z[zi];
  red[vi] = zv * zv;
  __syncthreads();
  for (int s = 64; s > 0; s >>= 1) {
    if (vi < s) red[vi] += red[vi + s];
    __syncthreads();
  }
  float rms  = zv * rsqrtf(red[0] / 128.f + EPS_);
  float gv   = g[(size_t)m * D_ + h * 128 + vi];
  float silu = gv / (1.f + expf(-gv));
  y[(size_t)m * D_ + h * 128 + vi] = (__bf16)(rms * normw[vi] * silu);
}

// ---------------------------------------------------------------------------
// Host orchestration
// ---------------------------------------------------------------------------
extern "C" void kernel_launch(void* const* d_in, const int* in_sizes, int n_in,
                              void* d_out, int out_size, void* d_ws,
                              size_t ws_size, hipStream_t stream) {
  (void)in_sizes; (void)n_in; (void)out_size; (void)ws_size;
  const float* X     = (const float*)d_in[0];
  const float* Wq    = (const float*)d_in[1];
  const float* Wk    = (const float*)d_in[2];
  const float* Wv    = (const float*)d_in[3];
  const float* Weta  = (const float*)d_in[4];
  const float* Wth   = (const float*)d_in[5];
  const float* Wg    = (const float*)d_in[6];
  const float* normw = (const float*)d_in[7];
  const float* Wo    = (const float*)d_in[8];

  size_t off = 0;
  auto carve = [&](size_t bytes) -> void* {
    void* p = (char*)d_ws + off;
    off += (bytes + 255) & ~(size_t)255;
    return p;
  };
  const size_t MT = (size_t)B_ * T_;  // 8192 rows

  __bf16* Xbf  = (__bf16*)carve(MT * D_ * 2);
  __bf16* Wqbf = (__bf16*)carve((size_t)D_ * D_ * 2);
  __bf16* Wkbf = (__bf16*)carve((size_t)D_ * D_ * 2);
  __bf16* Wvbf = (__bf16*)carve((size_t)D_ * D_ * 2);
  __bf16* Wgbf = (__bf16*)carve((size_t)D_ * D_ * 2);
  __bf16* Wobf = (__bf16*)carve((size_t)D_ * D_ * 2);
  float*  qf   = (float*)carve(MT * D_ * 4);  // [B,H,T,128]
  float*  kf   = (float*)carve(MT * D_ * 4);
  float*  vf   = (float*)carve(MT * D_ * 4);
  float*  gf   = (float*)carve(MT * D_ * 4);  // [M, 2048]
  float*  etaf = (float*)carve((size_t)B_ * H_ * T_ * 4);
  float*  thf  = (float*)carve((size_t)B_ * H_ * T_ * 4);
  float*  zf   = (float*)carve(MT * D_ * 4);  // [B,H,T,128]
  __bf16* ybf  = (__bf16*)carve(MT * D_ * 2);

  // 1) bf16 conversions
  cvt_f32_bf16<<<2048, 256, 0, stream>>>(X,  Xbf,  (long long)MT * D_);
  cvt_f32_bf16<<<1024, 256, 0, stream>>>(Wq, Wqbf, (long long)D_ * D_);
  cvt_f32_bf16<<<1024, 256, 0, stream>>>(Wk, Wkbf, (long long)D_ * D_);
  cvt_f32_bf16<<<1024, 256, 0, stream>>>(Wv, Wvbf, (long long)D_ * D_);
  cvt_f32_bf16<<<1024, 256, 0, stream>>>(Wg, Wgbf, (long long)D_ * D_);
  cvt_f32_bf16<<<1024, 256, 0, stream>>>(Wo, Wobf, (long long)D_ * D_);

  // 2) projections (WMMA bf16, f32 accumulate), 128x128 block tiles
  dim3 gg(D_ / 128, MT / 128);  // (16, 64)
  const float qscale = 1.0f / sqrtf((float)K_);
  gemm_bf16_128x128<1><<<gg, 256, 0, stream>>>(Xbf, Wqbf, qf, (int)MT, D_, D_, qscale);
  gemm_bf16_128x128<1><<<gg, 256, 0, stream>>>(Xbf, Wkbf, kf, (int)MT, D_, D_, 1.f);
  gemm_bf16_128x128<1><<<gg, 256, 0, stream>>>(Xbf, Wvbf, vf, (int)MT, D_, D_, 1.f);
  gemm_bf16_128x128<0><<<gg, 256, 0, stream>>>(Xbf, Wgbf, gf, (int)MT, D_, D_, 1.f);

  // 3) eta/theta (small-N projections, VALU)
  etatheta_kernel<<<(B_ * T_ * H_ + 255) / 256, 256, 0, stream>>>(X, Weta, Wth,
                                                                  etaf, thf);

  // 4) chunked fast-weight scan, LDS-resident state + swizzle scratch + TDM
  const size_t smem =
      (size_t)(128 * 128 + 4 * 64 * 128 + 64 * 64 + 4 * 64) * sizeof(float) +
      (size_t)2 * 4 * 128 * SWS_ * sizeof(__bf16);
  scan_kernel<<<B_ * H_, 256, smem, stream>>>(qf, kf, vf, etaf, thf, zf);

  // 5) gated RMS norm -> bf16
  normgate_kernel<<<B_ * T_ * H_, 128, 0, stream>>>(zf, gf, normw, ybf);

  // 6) output projection into d_out (f32)
  gemm_bf16_128x128<0><<<gg, 256, 0, stream>>>(ybf, Wobf, (float*)d_out, (int)MT,
                                               D_, D_, 1.f);
}
